// AttentionPruneKV_inference_70918499992292
// MI455X (gfx1250) — compile-verified, matrix operand order checked by reading, not run on hardware
//
#include <hip/hip_runtime.h>
#include <hip/hip_bf16.h>

// ---------------------------------------------------------------------------
// AttentionPruneKV for MI455X (gfx1250), wave32 + v_wmma_f32_16x16x32_f16.
// Inputs (fp32): x, prob, w_qkv, table1, table2, headsita, w_thresh, w_out, b_out
// Outputs (fp32): out [1,1600,512] then attn0 [1,8,1600,1600], concatenated.
//
// Roofline: mandatory HBM traffic is dominated by the attn0 output (82 MB,
// ~3.5us @ 23.3 TB/s). Total matmul work ~8.6 GFLOP; with f16 WMMA (fp32
// accum) compute lands near the memory floor, so all GEMMs use
// v_wmma_f32_16x16x32_f16 with 1x4 register blocking. Score->softmax->prune->AV
// is fused per 16-query block with the 16x1600 slab held in LDS (104 KB of the
// 320 KB/WGP CDNA5 budget), so the [H,N,N] attention tensors never hit HBM
// except the required attn0 output (written non-temporally).
// ---------------------------------------------------------------------------

typedef __attribute__((ext_vector_type(16))) _Float16 v16h;
typedef __attribute__((ext_vector_type(8)))  float    v8f;

#define H_   8
#define DH_  64
#define GH_  40
#define GW_  40
#define NP_  1600
#define DIM_ 512
#define QKVD_ 1536

// ---- WMMA wrapper ---------------------------------------------------------
__device__ inline v8f wmma16(v16h a, v16h b, v8f c) {
  // (neg_a, A, neg_b, B, c_mod, C, reuse_a, reuse_b)
  return __builtin_amdgcn_wmma_f32_16x16x32_f16(false, a, false, b,
                                                (short)0, c, false, false);
}

// ---- Fragment loaders (ISA 7.12.2: 16-bit A 16x32 layout) -----------------
// element e -> VGPR v=e/2, half e&1 ; k = (v&4?16:0) + (lane>=16?8:0) + (v&3)*2 + (e&1)
// A-style: element at base[m*ld + k]  (m = lane&15); contiguous in k -> b128 loads
__device__ inline v16h frag_mk(const _Float16* __restrict__ base, int ld, int k0, int lane) {
  v16h f;
  int m  = lane & 15;
  int kb = ((lane >> 4) & 1) * 8;
#pragma unroll
  for (int e = 0; e < 16; ++e) {
    int v = e >> 1;
    int k = ((v & 4) ? 16 : 0) + kb + ((v & 3) << 1) + (e & 1);
    f[e] = base[m * ld + k0 + k];
  }
  return f;
}
// A-style but from fp32 LDS (convert to f16 on load)
__device__ inline v16h frag_mk_f32(const float* __restrict__ base, int ld, int k0, int lane) {
  v16h f;
  int m  = lane & 15;
  int kb = ((lane >> 4) & 1) * 8;
#pragma unroll
  for (int e = 0; e < 16; ++e) {
    int v = e >> 1;
    int k = ((v & 4) ? 16 : 0) + kb + ((v & 3) << 1) + (e & 1);
    f[e] = (_Float16)base[m * ld + k0 + k];
  }
  return f;
}

// ---- fp32 -> fp16 conversion ----------------------------------------------
__global__ void f2h_kernel(const float* __restrict__ src, _Float16* __restrict__ dst, int n) {
  int i = blockIdx.x * blockDim.x + threadIdx.x;
  if (i < n) dst[i] = (_Float16)src[i];
}

// ---- Kernel A: QKV projection  C[1600,1536] = xh . w_qkv^T ----------------
// one wave per 16x64 tile (1x4 register blocking); scatter into
// q,k: [H][N][DH] ; v transposed: [H][DH][N] (so AV B-frags vectorize).
__global__ void qkv_gemm_kernel(const _Float16* __restrict__ xh,
                                const _Float16* __restrict__ wh,
                                _Float16* __restrict__ qkvh) {
  int lane = threadIdx.x & 31;
  int row0 = blockIdx.y * 16;
  int col0 = blockIdx.x * 64;
  const _Float16* a0 = xh + (size_t)row0 * DIM_;
  const _Float16* b0 = wh + (size_t)col0 * DIM_;  // B[k][n] = w_qkv[col0+n][k]
  v8f acc0 = {}, acc1 = {}, acc2 = {}, acc3 = {};
#pragma unroll 2
  for (int k0 = 0; k0 < DIM_; k0 += 32) {
    v16h a = frag_mk(a0, DIM_, k0, lane);
    v16h b0f = frag_mk(b0 + (size_t)0 * 16 * DIM_, DIM_, k0, lane);
    v16h b1f = frag_mk(b0 + (size_t)1 * 16 * DIM_, DIM_, k0, lane);
    v16h b2f = frag_mk(b0 + (size_t)2 * 16 * DIM_, DIM_, k0, lane);
    v16h b3f = frag_mk(b0 + (size_t)3 * 16 * DIM_, DIM_, k0, lane);
    acc0 = wmma16(a, b0f, acc0);
    acc1 = wmma16(a, b1f, acc1);
    acc2 = wmma16(a, b2f, acc2);
    acc3 = wmma16(a, b3f, acc3);
  }
  v8f accs[4] = {acc0, acc1, acc2, acc3};
#pragma unroll
  for (int t = 0; t < 4; ++t) {
#pragma unroll
    for (int r = 0; r < 8; ++r) {
      int m = row0 + r + ((lane >> 4) & 1) * 8;
      int j = col0 + t * 16 + (lane & 15);
      int which = j >> 9;          // 0=q 1=k 2=v
      int hh    = (j >> 6) & 7;
      int d     = j & 63;
      _Float16 hv = (_Float16)accs[t][r];
      if (which == 2)  // V stored transposed [H][DH][N]
        qkvh[(((size_t)(2 * H_ + hh) * DH_) + d) * NP_ + m] = hv;
      else
        qkvh[(((size_t)(which * H_ + hh) * NP_) + m) * DH_ + d] = hv;
    }
  }
}

// ---- Kernel B: fused attention per (h, 16-query block) --------------------
__global__ void attn_kernel(const _Float16* __restrict__ qkvh,
                            const float* __restrict__ table1,
                            const float* __restrict__ table2,
                            const float* __restrict__ headsita,
                            const float* __restrict__ w_thresh,
                            float* __restrict__ attn0_out,
                            _Float16* __restrict__ out0h) {
  extern __shared__ float smem[];
  float* S      = smem;              // [16][1600] score slab
  float* red    = S + 16 * NP_;      // [16][16] reduction scratch
  float* rowmax = red + 256;         // [16]
  float* rowsum = rowmax + 16;       // [16]
  float* rowmin = rowsum + 16;       // [16]
  float* rowth  = rowmin + 16;       // [16]
  float* rowfac = rowth + 16;        // [16]

  const int tid  = threadIdx.x;
  const int lane = tid & 31;
  const int wave = tid >> 5;
  const int h    = blockIdx.y;
  const int row0 = blockIdx.x * 16;
  const float scale = 0.125f;  // 64^-0.5

  const _Float16* qb  = qkvh + ((size_t)(0 * H_ + h) * NP_ + row0) * DH_;
  const _Float16* kbp = qkvh + ((size_t)(1 * H_ + h) * NP_) * DH_;
  const _Float16* vtb = qkvh + ((size_t)(2 * H_ + h) * DH_) * NP_;  // [DH][N]

  // ---- step 1: S0 = scale * Q K^T  (WMMA, 100 col-tiles over 8 waves)
  {
    v16h aq0 = frag_mk(qb, DH_, 0, lane);    // loop-invariant Q fragments
    v16h aq1 = frag_mk(qb, DH_, 32, lane);
    for (int ct = wave; ct < NP_ / 16; ct += 8) {
      int c0 = ct * 16;
      __builtin_prefetch((const void*)(kbp + (size_t)(c0 + 128) * DH_), 0, 1);
      v8f acc = {};
      v16h b0 = frag_mk(kbp + (size_t)c0 * DH_, DH_, 0, lane);
      acc = wmma16(aq0, b0, acc);
      v16h b1 = frag_mk(kbp + (size_t)c0 * DH_, DH_, 32, lane);
      acc = wmma16(aq1, b1, acc);
#pragma unroll
      for (int r = 0; r < 8; ++r) {
        int m = r + ((lane >> 4) & 1) * 8;
        S[m * NP_ + c0 + (lane & 15)] = acc[r] * scale;
      }
    }
  }
  __syncthreads();

  const int rrow = tid >> 4;   // 16 threads per row
  const int sub  = tid & 15;

  // ---- step 2: softmax stats of S0, write attn0 (non-temporal: write-once)
  {
    float lm = -3.4e38f;
    for (int c = sub; c < NP_; c += 16) lm = fmaxf(lm, S[rrow * NP_ + c]);
    red[rrow * 16 + sub] = lm;
    __syncthreads();
    if (tid < 16) {
      float m2 = -3.4e38f;
      for (int j = 0; j < 16; ++j) m2 = fmaxf(m2, red[tid * 16 + j]);
      rowmax[tid] = m2;
    }
    __syncthreads();
    float ls = 0.f, rm = rowmax[rrow];
    for (int c = sub; c < NP_; c += 16) ls += __expf(S[rrow * NP_ + c] - rm);
    red[rrow * 16 + sub] = ls;
    __syncthreads();
    if (tid < 16) {
      float s2 = 0.f;
      for (int j = 0; j < 16; ++j) s2 += red[tid * 16 + j];
      rowsum[tid] = s2;
    }
    __syncthreads();
    float iz = 1.0f / rowsum[rrow];
    float* dst = attn0_out + ((size_t)h * NP_ + row0 + rrow) * NP_;
    for (int c = sub; c < NP_; c += 16)
      __builtin_nontemporal_store(__expf(S[rrow * NP_ + c] - rm) * iz, &dst[c]);
  }
  __syncthreads();

  // ---- step 3: S += table1*table2 gathered bias + 0.01*exp(-factor*dis)
  {
    float sita   = headsita[h];
    float factor = 1.0f / (2.0f * sita * sita + 1e-6f);
    int qn = row0 + rrow;
    int ghq = qn / GW_, gwq = qn % GW_;
    for (int c = sub; c < NP_; c += 16) {
      int ghk = c / GW_, gwk = c % GW_;
      int dh_ = ghq - ghk, dw_ = gwq - gwk;
      float fh  = (float)dh_ * (1.0f / (float)GH_);
      float fw  = (float)dw_ * (1.0f / (float)GW_);
      float fw2 = (float)dw_ * (1.0f / 96.0f);
      float dis = fh * fh + fw * fw + fw2 * fw2;
      int idx = (dh_ + GH_ - 1) * (2 * GW_ - 1) + (dw_ + GW_ - 1);
      float bias = table1[idx * H_ + h] * table2[idx * H_ + h];
      S[rrow * NP_ + c] += bias + 0.01f * __expf(-factor * dis);
    }
  }
  __syncthreads();

  // ---- step 4: stats of biased S (max, min, sum-exp)
  {
    float lm = -3.4e38f, ln = 3.4e38f;
    for (int c = sub; c < NP_; c += 16) {
      float v = S[rrow * NP_ + c];
      lm = fmaxf(lm, v);
      ln = fminf(ln, v);
    }
    red[rrow * 16 + sub] = lm;
    __syncthreads();
    if (tid < 16) {
      float m2 = -3.4e38f;
      for (int j = 0; j < 16; ++j) m2 = fmaxf(m2, red[tid * 16 + j]);
      rowmax[tid] = m2;
    }
    __syncthreads();
    red[rrow * 16 + sub] = ln;
    __syncthreads();
    if (tid < 16) {
      float m2 = 3.4e38f;
      for (int j = 0; j < 16; ++j) m2 = fminf(m2, red[tid * 16 + j]);
      rowmin[tid] = m2;
    }
    __syncthreads();
    float ls = 0.f, rm = rowmax[rrow];
    for (int c = sub; c < NP_; c += 16) ls += __expf(S[rrow * NP_ + c] - rm);
    red[rrow * 16 + sub] = ls;
    __syncthreads();
    if (tid < 16) {
      float s2 = 0.f;
      for (int j = 0; j < 16; ++j) s2 += red[tid * 16 + j];
      rowsum[tid] = s2;
    }
    __syncthreads();
  }

  // ---- step 5: per-row threshold  thr = sigmoid(q.w_thresh)*sigmoid(-2)
  if (tid < 16) {
    float s = 0.f;
    const _Float16* qr = qb + (size_t)tid * DH_;
    for (int d = 0; d < DH_; ++d) s += (float)qr[d] * w_thresh[d];
    float sig  = 1.0f / (1.0f + __expf(-s));
    float amax = 1.0f / rowsum[tid];                               // exp(0)/Z
    float amin = __expf(rowmin[tid] - rowmax[tid]) / rowsum[tid];
    rowth[tid] = amin + sig * 0.11920292202211755f * (amax - amin);
  }
  __syncthreads();

  // ---- step 6: prune + renormalize (S becomes attn_p, fp32 in LDS)
  {
    float rm = rowmax[rrow], iz = 1.0f / rowsum[rrow], th = rowth[rrow];
    float ld = 0.f;
    for (int c = sub; c < NP_; c += 16) {
      float a   = __expf(S[rrow * NP_ + c] - rm) * iz;
      float val = (a > th) ? a : 0.0f;
      S[rrow * NP_ + c] = val;
      ld += val;
    }
    red[rrow * 16 + sub] = ld;
    __syncthreads();
    if (tid < 16) {
      float s2 = 0.f;
      for (int j = 0; j < 16; ++j) s2 += red[tid * 16 + j];
      rowfac[tid] = 1.0f / (s2 + 1e-6f);
    }
    __syncthreads();
    float fac = rowfac[rrow];
    for (int c = sub; c < NP_; c += 16) S[rrow * NP_ + c] *= fac;
  }
  __syncthreads();

  // ---- step 7: out0 = attn_p . V^T-layout  (8 waves: 4 col tiles x 2 K-halves)
  {
    int cw  = wave & 3;
    int kh2 = wave >> 2;
    int n0  = cw * 16;
    const _Float16* vb = vtb + (size_t)n0 * NP_;  // B[k][n] = vT[n0+n][k]
    v8f acc = {};
    for (int k0 = kh2 * 800; k0 < kh2 * 800 + 800; k0 += 32) {
      v16h a = frag_mk_f32(S, NP_, k0, lane);
      v16h b = frag_mk(vb, NP_, k0, lane);
      acc = wmma16(a, b, acc);
    }
    __syncthreads();                 // everyone done reading S
    if (wave >= 4) {                 // stash partial sums in (now free) S
#pragma unroll
      for (int r = 0; r < 8; ++r) S[(wave - 4) * 256 + r * 32 + lane] = acc[r];
    }
    __syncthreads();
    if (wave < 4) {
#pragma unroll
      for (int r = 0; r < 8; ++r) acc[r] += S[wave * 256 + r * 32 + lane];
#pragma unroll
      for (int r = 0; r < 8; ++r) {
        int m = r + ((lane >> 4) & 1) * 8;
        out0h[(size_t)(row0 + m) * DIM_ + h * DH_ + n0 + (lane & 15)] = (_Float16)acc[r];
      }
    }
  }
}

// ---- Kernel C: out = out0 . w_out^T + b_out, masked by sel ----------------
// 16x64 tile per wave (1x4 register blocking)
__global__ void out_gemm_kernel(const _Float16* __restrict__ out0h,
                                const _Float16* __restrict__ wouth,
                                const float* __restrict__ b_out,
                                const float* __restrict__ prob,
                                float* __restrict__ out) {
  int lane = threadIdx.x & 31;
  int row0 = blockIdx.y * 16;
  int col0 = blockIdx.x * 64;
  const _Float16* a0 = out0h + (size_t)row0 * DIM_;
  const _Float16* b0 = wouth + (size_t)col0 * DIM_;  // B[k][n] = w_out[col0+n][k]
  v8f acc0 = {}, acc1 = {}, acc2 = {}, acc3 = {};
#pragma unroll 2
  for (int k0 = 0; k0 < DIM_; k0 += 32) {
    v16h a = frag_mk(a0, DIM_, k0, lane);
    v16h b0f = frag_mk(b0 + (size_t)0 * 16 * DIM_, DIM_, k0, lane);
    v16h b1f = frag_mk(b0 + (size_t)1 * 16 * DIM_, DIM_, k0, lane);
    v16h b2f = frag_mk(b0 + (size_t)2 * 16 * DIM_, DIM_, k0, lane);
    v16h b3f = frag_mk(b0 + (size_t)3 * 16 * DIM_, DIM_, k0, lane);
    acc0 = wmma16(a, b0f, acc0);
    acc1 = wmma16(a, b1f, acc1);
    acc2 = wmma16(a, b2f, acc2);
    acc3 = wmma16(a, b3f, acc3);
  }
  v8f accs[4] = {acc0, acc1, acc2, acc3};
#pragma unroll
  for (int t = 0; t < 4; ++t) {
#pragma unroll
    for (int r = 0; r < 8; ++r) {
      int m = row0 + r + ((lane >> 4) & 1) * 8;
      int c = col0 + t * 16 + (lane & 15);
      float v = accs[t][r] + b_out[c];
      bool sel = prob[m] < 0.9f;
      out[(size_t)m * DIM_ + c] = sel ? v : 0.0f;
    }
  }
}

// ---------------------------------------------------------------------------
extern "C" void kernel_launch(void* const* d_in, const int* in_sizes, int n_in,
                              void* d_out, int out_size, void* d_ws, size_t ws_size,
                              hipStream_t stream) {
  const float* x        = (const float*)d_in[0];
  const float* prob     = (const float*)d_in[1];
  const float* w_qkv    = (const float*)d_in[2];
  const float* table1   = (const float*)d_in[3];
  const float* table2   = (const float*)d_in[4];
  const float* headsita = (const float*)d_in[5];
  const float* w_thresh = (const float*)d_in[6];
  const float* w_out    = (const float*)d_in[7];
  const float* b_out    = (const float*)d_in[8];

  float* out   = (float*)d_out;
  float* attn0 = out + (size_t)NP_ * DIM_;   // second tuple element

  // workspace (fp16 staging): ~10.3 MB
  _Float16* xh    = (_Float16*)d_ws;
  _Float16* wqkvh = xh + (size_t)NP_ * DIM_;
  _Float16* wouth = wqkvh + (size_t)QKVD_ * DIM_;
  _Float16* qkvh  = wouth + (size_t)DIM_ * DIM_;
  _Float16* out0h = qkvh + (size_t)3 * H_ * NP_ * DH_;

  f2h_kernel<<<(NP_ * DIM_ + 255) / 256, 256, 0, stream>>>(x, xh, NP_ * DIM_);
  f2h_kernel<<<(QKVD_ * DIM_ + 255) / 256, 256, 0, stream>>>(w_qkv, wqkvh, QKVD_ * DIM_);
  f2h_kernel<<<(DIM_ * DIM_ + 255) / 256, 256, 0, stream>>>(w_out, wouth, DIM_ * DIM_);

  qkv_gemm_kernel<<<dim3(QKVD_ / 64, NP_ / 16), 32, 0, stream>>>(xh, wqkvh, qkvh);

  size_t lds_bytes = (size_t)(16 * NP_ + 256 + 16 * 5) * sizeof(float);  // ~104 KB (CDNA5: 320 KB/WGP)
  attn_kernel<<<dim3(NP_ / 16, H_), 256, lds_bytes, stream>>>(
      qkvh, table1, table2, headsita, w_thresh, attn0, out0h);

  out_gemm_kernel<<<dim3(DIM_ / 64, NP_ / 16), 32, 0, stream>>>(out0h, wouth, b_out, prob, out);
}
